// DeconBlock_71605694759655
// MI455X (gfx1250) — matrix operand order repaired and checked by reading, not compiled, not run on hardware
//
#include <hip/hip_runtime.h>
#include <math.h>

// Problem constants (fixed by the reference).
#define BB   4
#define CCH  128
#define HH   128
#define WW   128
#define HWSZ (HH * WW)
#define EPSV 1e-5f

typedef __attribute__((ext_vector_type(16))) __bf16 v16bf;
typedef __attribute__((ext_vector_type(8)))  float  v8f;

union BfVec { v16bf v; unsigned int u[8]; };

__device__ __forceinline__ unsigned short f2bf(float f) {
    unsigned int x = __float_as_uint(f);
    x += 0x7FFFu + ((x >> 16) & 1u);   // round-to-nearest-even
    return (unsigned short)(x >> 16);
}

// LDS byte-offset of a __shared__ object (addrspace(3) cast -> group offset).
#define LDS_OFF(p) ((unsigned)(unsigned long long)(__attribute__((address_space(3))) const void*)(p))

// Async copy of one packed-bf16 weight chunk (contiguous `bytes` at src) into
// the padded As layout: row = 32 ushorts (64B) padded to 72B. Each thread moves
// 16B per issue; 16B segments never cross a padded row. Tracked by ASYNCcnt.
__device__ __forceinline__ void async_copy_wchunk(const unsigned short* src,
                                                  const unsigned short* As,
                                                  int t, int bytes) {
    unsigned lds = LDS_OFF(As) + (unsigned)((t >> 2) * 72 + (t & 3) * 16);
    unsigned long long gbase = (unsigned long long)src;
    int nIss = bytes >> 12;                     // 256 threads * 16B = 4096B/issue
    for (int it = 0; it < nIss; ++it) {
        unsigned voff = (unsigned)(t * 16 + it * 4096);
        unsigned vlds = lds + (unsigned)(it * 4608);   // 64 rows * 72B
        asm volatile("global_load_async_to_lds_b128 %0, %1, %2"
                     :: "v"(vlds), "v"(voff), "s"(gbase) : "memory");
    }
    asm volatile("s_wait_asynccnt 0x0" ::: "memory");
}

// ---------------------------------------------------------------------------
// Weight packing: wp[(kc*Opad + o)*32 + kk] = bf16(W[o][k]) with k = kc*32+kk.
// 3x3 conv K-order: k = kidx*128 + c  (kidx = ky*3+kx constant per chunk).
// ---------------------------------------------------------------------------
__global__ void pack3x3_k(const float* __restrict__ w, unsigned short* __restrict__ wp,
                          int Oreal, int Opad) {
    int idx = blockIdx.x * 256 + threadIdx.x;
    int tot = 36 * Opad * 32;
    if (idx >= tot) return;
    int kk  = idx & 31;
    int tmp = idx >> 5;
    int o   = tmp % Opad;
    int kc  = tmp / Opad;
    int kidx = kc >> 2;
    int c    = ((kc & 3) << 5) + kk;
    float v = (o < Oreal) ? w[(o * CCH + c) * 9 + kidx] : 0.f;
    wp[idx] = f2bf(v);
}

// Deformable conv K-order: k = ((d*9 + kidx)*64 + cg), chunk = half a group.
__global__ void packdef_k(const float* __restrict__ w, unsigned short* __restrict__ wp) {
    int idx = blockIdx.x * 256 + threadIdx.x;
    if (idx >= 36 * 128 * 32) return;
    int kk = idx & 31;
    int o  = (idx >> 5) & 127;
    int kc = idx >> 12;
    int k    = kc * 32 + kk;
    int d    = k / 576;
    int rem  = k - d * 576;
    int kidx = rem >> 6;
    int cg   = rem & 63;
    int c    = d * 64 + cg;
    wp[idx] = f2bf(w[(o * CCH + c) * 9 + kidx]);
}

__global__ void pack1x1_k(const float* __restrict__ w, unsigned short* __restrict__ wp) {
    int idx = blockIdx.x * 256 + threadIdx.x;
    if (idx >= 4 * 128 * 32) return;
    int kk = idx & 31;
    int o  = (idx >> 5) & 127;
    int kc = idx >> 12;
    int c  = kc * 32 + kk;
    wp[idx] = f2bf(w[o * CCH + c]);
}

// Fold conv-bias + eval-mode BN into per-channel affine: y = acc*scale + shift.
__global__ void prep_affine_k(const float* __restrict__ bn, const float* __restrict__ bias,
                              float* __restrict__ scale, float* __restrict__ shift, int Creal) {
    int o = threadIdx.x;
    float sc = 0.f, sh = 0.f;
    if (o < Creal) {
        if (bn) {
            float g  = bn[o];
            float be = bn[128 + o];
            float mu = bn[256 + o];
            float va = bn[384 + o];
            sc = g * rsqrtf(va + EPSV);
            sh = be - mu * sc;
            if (bias) sh += bias[o] * sc;
        } else {
            sc = 1.f;
            sh = bias ? bias[o] : 0.f;
        }
    }
    scale[o] = sc;
    shift[o] = sh;
}

// ---------------------------------------------------------------------------
// WMMA fragment loads from LDS (contiguous k order, same in A and B).
// A (16x32 bf16): lane l: M = l&15; K base = (l>>4)*16, packed 2/VGPR.
// B (32x16 bf16): lane l: N = l&15; K base = (l>>4)*16, packed 2/VGPR.
// ---------------------------------------------------------------------------
__device__ __forceinline__ void load_frag_a(BfVec& a, const unsigned short* Ar, int half) {
#pragma unroll
    for (int j = 0; j < 8; ++j)
        a.u[j] = *(const unsigned int*)&Ar[half * 16 + 2 * j];
}
__device__ __forceinline__ void load_frag_b(BfVec& b, const unsigned short* Br) {
#pragma unroll
    for (int j = 0; j < 8; ++j)
        b.u[j] = *(const unsigned int*)&Br[2 * j];
}

// ---------------------------------------------------------------------------
// Implicit-GEMM conv via v_wmma_f32_16x16x32_bf16.
// Block: 256 threads = 8 waves; tile = [MT*16 out-ch] x [64 pixels].
// Wave w: M-pair (w&3) -> out-ch [32*(w&3), +32); N-pair (w>>2) -> px 32*(w>>2)+{0,16}.
// act: 0=none, 1=relu, 2=leaky_relu(0.1). res_post: add residual after act.
// ---------------------------------------------------------------------------
__global__ __launch_bounds__(256)
void conv_wmma_k(const float* __restrict__ in, const unsigned short* __restrict__ wp,
                 const float* __restrict__ scale, const float* __restrict__ shift,
                 const float* __restrict__ res, float* __restrict__ out,
                 int nkc, int is3x3, int MT, int Cout, int act, int res_post) {
    __shared__ unsigned short As[128 * 36];   // [o][kk] padded rows (bank-skew)
    __shared__ unsigned short Bs[64 * 36];    // [pixel][kk]

    const int pix = blockIdx.x * 64;          // 64 consecutive pixels, same row
    const int b   = pix >> 14;
    const int y   = (pix >> 7) & (HH - 1);
    const int w0  = pix & (WW - 1);

    const int t    = threadIdx.x;
    const int wv   = t >> 5;
    const int l    = t & 31;
    const int half = l >> 4;
    const int nn   = l & 15;
    const int mp   = wv & 3;                  // M-pair
    const int np   = wv >> 2;                 // N-pair
    const int atot = MT * 512;                // ushorts per weight chunk
    const bool active = (2 * mp) < MT;

    v8f acc00 = {}, acc01 = {}, acc10 = {}, acc11 = {};

    for (int kc = 0; kc < nkc; ++kc) {
        __syncthreads();
        // ---- async DMA of A tile (packed bf16 weights) into LDS ----
        async_copy_wchunk(wp + (size_t)kc * atot, As, t, atot * 2);
        // ---- stage B tile: 32 channels x 64 pixels, bf16-converted ----
        {
            int dy = 0, dx = 0, c0;
            if (is3x3) {
                int kidx = kc >> 2;
                dy = kidx / 3 - 1;
                dx = kidx % 3 - 1;
                c0 = (kc & 3) << 5;
            } else {
                c0 = kc << 5;
            }
            int n  = t & 63;
            int ci = t >> 6;                   // 0..3 -> 8 channels each
            int yy = y + dy;
            int xx = w0 + n + dx;
            bool ok = (yy >= 0) && (yy < HH) && (xx >= 0) && (xx < WW);
            #pragma unroll
            for (int r = 0; r < 8; ++r) {
                int cc = ci * 8 + r;
                float v = 0.f;
                if (ok) v = in[((b * CCH + c0 + cc) * HH + yy) * WW + xx];
                Bs[n * 36 + cc] = f2bf(v);
            }
        }
        __syncthreads();
        // ---- 2x2 WMMA tile per wave ----
        if (active) {
            BfVec a0, a1, b0, b1;
            load_frag_a(a0, &As[((2 * mp) * 16 + nn) * 36], half);
            load_frag_a(a1, &As[((2 * mp + 1) * 16 + nn) * 36], half);
            const unsigned short* Br = &Bs[(np * 32 + nn) * 36 + half * 16];
            load_frag_b(b0, Br);
            load_frag_b(b1, Br + 16 * 36);
            acc00 = __builtin_amdgcn_wmma_f32_16x16x32_bf16(false, a0.v, false, b0.v,
                                                            (short)0, acc00, false, false);
            acc01 = __builtin_amdgcn_wmma_f32_16x16x32_bf16(false, a0.v, false, b1.v,
                                                            (short)0, acc01, false, false);
            acc10 = __builtin_amdgcn_wmma_f32_16x16x32_bf16(false, a1.v, false, b0.v,
                                                            (short)0, acc10, false, false);
            acc11 = __builtin_amdgcn_wmma_f32_16x16x32_bf16(false, a1.v, false, b1.v,
                                                            (short)0, acc11, false, false);
        }
    }

    // ---- epilogue: affine + residual + activation ----
    if (active) {
        v8f accs[2][2] = {{acc00, acc01}, {acc10, acc11}};
        #pragma unroll
        for (int mi = 0; mi < 2; ++mi) {
            #pragma unroll
            for (int r = 0; r < 8; ++r) {
                int o  = (2 * mp + mi) * 16 + half * 8 + r;
                float s  = scale[o];
                float sh = shift[o];
                int ob = ((b * Cout + o) * HH + y) * WW;
                int rb = ((b * CCH + o) * HH + y) * WW;
                #pragma unroll
                for (int ni = 0; ni < 2; ++ni) {
                    int col = w0 + np * 32 + ni * 16 + nn;
                    float v = accs[mi][ni][r] * s + sh;
                    if (res && !res_post) v += res[rb + col];
                    if (act == 1)      v = fmaxf(v, 0.f);
                    else if (act == 2) v = v > 0.f ? v : 0.1f * v;
                    if (res && res_post) v += res[rb + col];
                    out[ob + col] = v;
                }
            }
        }
    }
}

// ---------------------------------------------------------------------------
// Modulated deformable conv as implicit GEMM: B-tile is the bilinear-gathered,
// mask-modulated sample matrix (built on the fly); A is packed dc_w.
// Epilogue: leaky_relu(bn1(acc + dc_b)) (folded into scale/shift).
// ---------------------------------------------------------------------------
__global__ __launch_bounds__(256)
void deform_wmma_k(const float* __restrict__ xin, const float* __restrict__ off,
                   const unsigned short* __restrict__ wp,
                   const float* __restrict__ scale, const float* __restrict__ shift,
                   float* __restrict__ out) {
    __shared__ unsigned short As[128 * 36];
    __shared__ unsigned short Bs[64 * 36];

    const int pix = blockIdx.x * 64;
    const int b   = pix >> 14;
    const int y   = (pix >> 7) & (HH - 1);
    const int w0  = pix & (WW - 1);

    const int t    = threadIdx.x;
    const int wv   = t >> 5;
    const int l    = t & 31;
    const int half = l >> 4;
    const int nn   = l & 15;
    const int mp   = wv & 3;
    const int np   = wv >> 2;

    v8f acc00 = {}, acc01 = {}, acc10 = {}, acc11 = {};

    for (int kc = 0; kc < 36; ++kc) {
        __syncthreads();
        // ---- async DMA of A tile into LDS ----
        async_copy_wchunk(wp + (size_t)kc * 4096, As, t, 8192);
        // ---- gather B tile: chunk = 32 channels of deform-group d, tap kidx ----
        {
            int dk   = kc >> 1;
            int d    = dk / 9;
            int kidx = dk - d * 9;
            int cg0  = (kc & 1) << 5;
            int n    = t & 63;
            int ci   = t >> 6;
            int col  = w0 + n;

            const float* ob = off + ((b * 64) * HH + y) * WW + col;
            int chy = (d * 9 + kidx) * 2;
            float oy = ob[chy * HWSZ];
            float ox = ob[(chy + 1) * HWSZ];
            float om = ob[(36 + d * 9 + kidx) * HWSZ];
            float mk = 1.f / (1.f + expf(-om));

            float py = oy + (float)y + (float)(kidx / 3 - 1);
            float px = ox + (float)col + (float)(kidx % 3 - 1);
            float fy0 = floorf(py), fx0 = floorf(px);
            int y0 = (int)fy0, x0 = (int)fx0;
            float fy = py - fy0, fx = px - fx0;
            float w00 = (1.f - fy) * (1.f - fx);
            float w01 = (1.f - fy) * fx;
            float w10 = fy * (1.f - fx);
            float w11 = fy * fx;
            bool vy0 = (y0 >= 0) && (y0 < HH);
            bool vy1 = (y0 + 1 >= 0) && (y0 + 1 < HH);
            bool vx0 = (x0 >= 0) && (x0 < WW);
            bool vx1 = (x0 + 1 >= 0) && (x0 + 1 < WW);
            int yc0 = min(max(y0, 0), HH - 1);
            int yc1 = min(max(y0 + 1, 0), HH - 1);
            int xc0 = min(max(x0, 0), WW - 1);
            int xc1 = min(max(x0 + 1, 0), WW - 1);
            float m00 = (vy0 && vx0) ? w00 * mk : 0.f;
            float m01 = (vy0 && vx1) ? w01 * mk : 0.f;
            float m10 = (vy1 && vx0) ? w10 * mk : 0.f;
            float m11 = (vy1 && vx1) ? w11 * mk : 0.f;

            #pragma unroll
            for (int r = 0; r < 8; ++r) {
                int cc = ci * 8 + r;
                const float* xp = xin + (size_t)(b * CCH + d * 64 + cg0 + cc) * HWSZ;
                float v = m00 * xp[yc0 * WW + xc0] + m01 * xp[yc0 * WW + xc1]
                        + m10 * xp[yc1 * WW + xc0] + m11 * xp[yc1 * WW + xc1];
                Bs[n * 36 + cc] = f2bf(v);
            }
        }
        __syncthreads();
        {
            BfVec a0, a1, b0, b1;
            load_frag_a(a0, &As[((2 * mp) * 16 + nn) * 36], half);
            load_frag_a(a1, &As[((2 * mp + 1) * 16 + nn) * 36], half);
            const unsigned short* Br = &Bs[(np * 32 + nn) * 36 + half * 16];
            load_frag_b(b0, Br);
            load_frag_b(b1, Br + 16 * 36);
            acc00 = __builtin_amdgcn_wmma_f32_16x16x32_bf16(false, a0.v, false, b0.v,
                                                            (short)0, acc00, false, false);
            acc01 = __builtin_amdgcn_wmma_f32_16x16x32_bf16(false, a0.v, false, b1.v,
                                                            (short)0, acc01, false, false);
            acc10 = __builtin_amdgcn_wmma_f32_16x16x32_bf16(false, a1.v, false, b0.v,
                                                            (short)0, acc10, false, false);
            acc11 = __builtin_amdgcn_wmma_f32_16x16x32_bf16(false, a1.v, false, b1.v,
                                                            (short)0, acc11, false, false);
        }
    }

    v8f accs[2][2] = {{acc00, acc01}, {acc10, acc11}};
    #pragma unroll
    for (int mi = 0; mi < 2; ++mi) {
        #pragma unroll
        for (int r = 0; r < 8; ++r) {
            int o  = (2 * mp + mi) * 16 + half * 8 + r;
            float s  = scale[o];
            float sh = shift[o];
            int ob = ((b * CCH + o) * HH + y) * WW;
            #pragma unroll
            for (int ni = 0; ni < 2; ++ni) {
                int col = w0 + np * 32 + ni * 16 + nn;
                float v = accs[mi][ni][r] * s + sh;
                v = v > 0.f ? v : 0.1f * v;   // leaky_relu(0.1)
                out[ob + col] = v;
            }
        }
    }
}

// ---------------------------------------------------------------------------
extern "C" void kernel_launch(void* const* d_in, const int* in_sizes, int n_in,
                              void* d_out, int out_size, void* d_ws, size_t ws_size,
                              hipStream_t stream) {
    (void)in_sizes; (void)n_in; (void)out_size; (void)ws_size;
    const float* x      = (const float*)d_in[0];
    const float* rb_w1  = (const float*)d_in[1];
    const float* rb_b1  = (const float*)d_in[2];
    const float* rb_bn1 = (const float*)d_in[3];
    const float* rb_w2  = (const float*)d_in[4];
    const float* rb_b2  = (const float*)d_in[5];
    const float* rb_bn2 = (const float*)d_in[6];
    const float* off_w  = (const float*)d_in[7];
    const float* off_b  = (const float*)d_in[8];
    const float* dc_w   = (const float*)d_in[9];
    const float* dc_b   = (const float*)d_in[10];
    const float* bn1    = (const float*)d_in[11];
    const float* cv2_w  = (const float*)d_in[12];
    const float* bn2    = (const float*)d_in[13];
    float* out = (float*)d_out;

    const size_t TSZ = (size_t)BB * CCH * HH * WW * sizeof(float);   // 32 MB
    char* p = (char*)d_ws;
    float* t1   = (float*)p; p += TSZ;
    float* t2   = (float*)p; p += TSZ;
    float* t3   = (float*)p; p += TSZ;
    float* offo = (float*)p; p += (size_t)BB * 64 * HH * WW * sizeof(float);
    unsigned short* wp1 = (unsigned short*)p; p += 36 * 128 * 32 * 2;
    unsigned short* wp2 = (unsigned short*)p; p += 36 * 128 * 32 * 2;
    unsigned short* wpd = (unsigned short*)p; p += 36 * 128 * 32 * 2;
    unsigned short* wpo = (unsigned short*)p; p += 36 * 64 * 32 * 2;
    unsigned short* wpc = (unsigned short*)p; p += 4 * 128 * 32 * 2;
    float* ss = (float*)p;   // 5 x (scale[128], shift[128])

    pack3x3_k<<<(36 * 128 * 32 + 255) / 256, 256, 0, stream>>>(rb_w1, wp1, 128, 128);
    pack3x3_k<<<(36 * 128 * 32 + 255) / 256, 256, 0, stream>>>(rb_w2, wp2, 128, 128);
    pack3x3_k<<<(36 * 64 * 32 + 255) / 256, 256, 0, stream>>>(off_w, wpo, 54, 64);
    packdef_k<<<(36 * 128 * 32 + 255) / 256, 256, 0, stream>>>(dc_w, wpd);
    pack1x1_k<<<(4 * 128 * 32 + 255) / 256, 256, 0, stream>>>(cv2_w, wpc);

    prep_affine_k<<<1, 128, 0, stream>>>(rb_bn1, rb_b1, ss + 0,    ss + 128,  128);
    prep_affine_k<<<1, 128, 0, stream>>>(rb_bn2, rb_b2, ss + 256,  ss + 384,  128);
    prep_affine_k<<<1, 128, 0, stream>>>(nullptr, off_b, ss + 512, ss + 640,  54);
    prep_affine_k<<<1, 128, 0, stream>>>(bn1,    dc_b,  ss + 768,  ss + 896,  128);
    prep_affine_k<<<1, 128, 0, stream>>>(bn2, nullptr,  ss + 1024, ss + 1152, 128);

    const int NB = BB * HH * WW / 64;   // 1024 blocks x 256 threads
    // ResBlock conv1: relu(bn(conv(x)+b))
    conv_wmma_k<<<NB, 256, 0, stream>>>(x,  wp1, ss + 0,   ss + 128, nullptr, t1,
                                        36, 1, 8, 128, 1, 0);
    // ResBlock conv2 + identity: relu(bn(conv(t1)+b) + x)
    conv_wmma_k<<<NB, 256, 0, stream>>>(t1, wp2, ss + 256, ss + 384, x, t2,
                                        36, 1, 8, 128, 1, 0);
    // Offset/mask conv (54 real ch padded to 64, linear epilogue)
    conv_wmma_k<<<NB, 256, 0, stream>>>(t2, wpo, ss + 512, ss + 640, nullptr, offo,
                                        36, 1, 4, 64, 0, 0);
    // Modulated deformable conv + bn1 + leaky_relu
    deform_wmma_k<<<NB, 256, 0, stream>>>(t2, offo, wpd, ss + 768, ss + 896, t3);
    // 1x1 conv + bn2 + leaky_relu, then final residual add of x (post-act)
    conv_wmma_k<<<NB, 256, 0, stream>>>(t3, wpc, ss + 1024, ss + 1152, x, out,
                                        4, 0, 8, 128, 2, 1);
}
